// Involution_76596446756896
// MI455X (gfx1250) — matrix operand order repaired and checked by reading, not compiled, not run on hardware
//
#include <hip/hip_runtime.h>
#include <hip/hip_bf16.h>

// ---------------------------------------------------------------------------
// Involution (B=4,H=56,W=56,C=256, G=16, K=7, r=4) for gfx1250 (MI455X).
// Stage 1: x @ w_reduce  + BN + ReLU      (bf16 WMMA, f32 accumulate)
// Stage 2: r @ w_span    + b_span         (bf16 WMMA, f32 accumulate,
//                                          async-to-LDS weight staging)
// Stage 3: per-pixel per-group 49-tap weighted gather (VALU, float4)
// ---------------------------------------------------------------------------

typedef __bf16 bf16;
typedef __bf16 v16bf __attribute__((ext_vector_type(16)));
typedef float  v8f   __attribute__((ext_vector_type(8)));
typedef float  v4f   __attribute__((ext_vector_type(4)));

#define BB 4
#define HH 56
#define WW 56
#define CC 256
#define CR 64
#define GG 16
#define CG 16
#define KS 7
#define KK 49
#define EE (KK * GG)      /* 784 */
#define PP (BB * HH * WW) /* 12544 pixels */

// --- A-fragment loader: 16x32 bf16 A tile, row-major source, per ISA layout.
// lane half h = lane>>4 ; elems 0..7 -> K = k0+8h .. k0+8h+7
//                         elems 8..15 -> K = k0+16+8h .. k0+16+8h+7
__device__ inline v16bf load_a_bf16(const bf16* row, int k0, int h) {
    v16bf a;
    const int b0 = k0 + 8 * h;
#pragma unroll
    for (int i = 0; i < 8; ++i) a[i] = row[b0 + i];
#pragma unroll
    for (int i = 0; i < 8; ++i) a[8 + i] = row[b0 + 16 + i];
    return a;
}
__device__ inline v16bf load_a_f32(const float* row, int k0, int h) {
    v16bf a;
    const int b0 = k0 + 8 * h;
#pragma unroll
    for (int i = 0; i < 8; ++i) a[i] = (bf16)row[b0 + i];
#pragma unroll
    for (int i = 0; i < 8; ++i) a[8 + i] = (bf16)row[b0 + 16 + i];
    return a;
}

// --- B-fragment loader: 32x16 bf16 B tile from LDS staged N-major
// (K contiguous, padded stride). lane n = lane&15, elems 0..7 -> K = k0+8h+i,
// elems 8..15 -> K = k0+16+8h+(i-8).
__device__ inline v16bf load_b_lds(const bf16* col, int k0, int h) {
    v16bf b;
    const int b0 = k0 + 8 * h;
#pragma unroll
    for (int i = 0; i < 8; ++i) b[i] = col[b0 + i];
#pragma unroll
    for (int i = 0; i < 8; ++i) b[8 + i] = col[b0 + 16 + i];
    return b;
}

// ===========================================================================
// Kernel 1: r = relu(BN(x @ w_reduce))   -> r_ws (bf16, [P][64])
// block = 256 threads (8 waves); each wave: one 16-row M tile, all 64 cols.
// ===========================================================================
#define LDSW_STRIDE 264   /* 256 + 8 pad : de-conflicts 64 banks */

__global__ __launch_bounds__(256)
void involution_reduce_bn(const float* __restrict__ x,
                          const float* __restrict__ w_reduce,
                          const float* __restrict__ gamma,
                          const float* __restrict__ beta,
                          const float* __restrict__ mean,
                          const float* __restrict__ var,
                          bf16* __restrict__ r_out) {
    __shared__ bf16 ldsW[CR * LDSW_STRIDE];   // w_reduce transposed: [d][c]

    const int tid = threadIdx.x;
    // Stage w_reduce (256x64 f32) -> LDS bf16, transposed to N-major.
    for (int idx = tid; idx < CC * CR; idx += 256) {
        const int c = idx >> 6;        // coalesced global read
        const int d = idx & 63;
        ldsW[d * LDSW_STRIDE + c] = (bf16)w_reduce[idx];
    }
    __syncthreads();

    const int wave = tid >> 5;
    const int lane = tid & 31;
    const int h    = lane >> 4;
    const int nl   = lane & 15;
    const int m0   = (blockIdx.x * 8 + wave) * 16;

    v8f acc[4];
#pragma unroll
    for (int t = 0; t < 4; ++t) acc[t] = (v8f)0.0f;

    const float* xrow = x + (size_t)(m0 + nl) * CC;

    for (int k0 = 0; k0 < CC; k0 += 32) {
        v16bf a = load_a_f32(xrow, k0, h);
#pragma unroll
        for (int t = 0; t < 4; ++t) {
            v16bf b = load_b_lds(&ldsW[(t * 16 + nl) * LDSW_STRIDE], k0, h);
            acc[t] = __builtin_amdgcn_wmma_f32_16x16x32_bf16(
                false, a, false, b, (short)0, acc[t], false, false);
        }
    }

    // Epilogue: BN (inference) + ReLU, store bf16.
#pragma unroll
    for (int t = 0; t < 4; ++t) {
        const int d = t * 16 + nl;
        const float sc = gamma[d] * rsqrtf(var[d] + 1e-3f);
        const float bi = beta[d] - mean[d] * sc;
#pragma unroll
        for (int i = 0; i < 8; ++i) {
            const int row = m0 + i + 8 * h;
            float v = fmaf(acc[t][i], sc, bi);
            v = fmaxf(v, 0.0f);
            r_out[(size_t)row * CR + d] = (bf16)v;
        }
    }
}

// ===========================================================================
// Kernel 2: kgen = r @ w_span + b_span  -> kgen_ws (f32, [P][784])
// grid (98, 7); block = 224 threads (7 waves). The 64x112 f32 slice of
// w_span is DMA'd into LDS with GLOBAL_LOAD_ASYNC_TO_LDS_B128 (ASYNCcnt),
// then converted LDS->LDS into an N-major bf16 tile. Wave w owns N-tile
// (by*7+w) and loops over 8 M-tiles.
// ===========================================================================
#define LDSS_STRIDE 72    /* 64 + 8 pad */

__global__ __launch_bounds__(224)
void involution_span(const bf16*  __restrict__ r_in,
                     const float* __restrict__ w_span,
                     const float* __restrict__ b_span,
                     float* __restrict__ kgen) {
    __shared__ float ldsF[CR * 112];          // async-copy target, [k][112] f32
    __shared__ bf16  ldsS[112 * LDSS_STRIDE]; // [n_local][k], n-major bf16

    const int tid   = threadIdx.x;
    const int nbase = blockIdx.y * 112;

    // ---- Async DMA: global w_span slice -> LDS (no VGPR round-trip). ----
    // Each lane copies distinct 16B chunks; 112/4=28 b128 ops per k row, so a
    // chunk never crosses a row. GVS addressing: lds-addr VGPR, 32-bit byte
    // offset VGPR, 64-bit SGPR base.
    {
        const unsigned ldsBase = (unsigned)(uintptr_t)&ldsF[0];
        for (int q = tid; q < (CR * 112) / 4; q += 224) {
            const int k  = q / 28;
            const int nn = (q - k * 28) * 4;
            const unsigned goff = (unsigned)(((size_t)k * EE + nbase + nn) * sizeof(float));
            const unsigned loff = ldsBase + (unsigned)q * 16u;
            asm volatile("global_load_async_to_lds_b128 %0, %1, %2"
                         :: "v"(loff), "v"(goff), "s"(w_span)
                         : "memory");
        }
        asm volatile("s_wait_asynccnt 0x0" ::: "memory");
    }
    __syncthreads();

    // ---- LDS->LDS: transpose to N-major and convert f32 -> bf16. ----
    for (int idx = tid; idx < 112 * CR; idx += 224) {
        const int nn = idx >> 6;
        const int k  = idx & 63;
        ldsS[nn * LDSS_STRIDE + k] = (bf16)ldsF[k * 112 + nn];
    }
    __syncthreads();

    const int wave = tid >> 5;
    const int lane = tid & 31;
    const int h    = lane >> 4;
    const int nl   = lane & 15;
    const int n    = nbase + wave * 16 + nl;            // global output column
    const bf16* bcol = &ldsS[(wave * 16 + nl) * LDSS_STRIDE];
    const float bias = b_span[n];

    for (int mt = 0; mt < 8; ++mt) {
        const int m0 = (blockIdx.x * 8 + mt) * 16;
        const bf16* arow = r_in + (size_t)(m0 + nl) * CR;

        v8f acc = (v8f)0.0f;
#pragma unroll
        for (int k0 = 0; k0 < CR; k0 += 32) {
            v16bf a = load_a_bf16(arow, k0, h);
            v16bf b = load_b_lds(bcol, k0, h);
            acc = __builtin_amdgcn_wmma_f32_16x16x32_bf16(
                false, a, false, b, (short)0, acc, false, false);
        }
#pragma unroll
        for (int i = 0; i < 8; ++i) {
            const int row = m0 + i + 8 * h;
            kgen[(size_t)row * EE + n] = acc[i] + bias;
        }
    }
}

// ===========================================================================
// Kernel 3: out[p, g*16+c] = sum_kk kgen[p, g*49+kk] * x[nbr(p,kk), g*16+c]
// One thread per (pixel, group): 4 x float4 accumulators (16 channels).
// Lane layout: g = t&15 fastest -> a wave covers 2 pixels x 256 contiguous
// channels (cache-line friendly float4 gathers).
// ===========================================================================
__global__ __launch_bounds__(256)
void involution_apply(const float* __restrict__ x,
                      const float* __restrict__ kgen,
                      float* __restrict__ out) {
    const int t = blockIdx.x * 256 + threadIdx.x;   // 0 .. P*G-1
    const int g = t & 15;
    const int p = t >> 4;
    const int b  = p / (HH * WW);
    const int hw = p - b * (HH * WW);
    const int hh = hw / WW;
    const int ww = hw - hh * WW;

    const float* wk = kgen + (size_t)p * EE + g * KK;

    v4f acc0 = (v4f)0.0f, acc1 = (v4f)0.0f, acc2 = (v4f)0.0f, acc3 = (v4f)0.0f;

    for (int i = 0; i < KS; ++i) {
        const int yh = hh + i - 3;
        if ((unsigned)yh >= HH) continue;          // zero padding row
        const float* rowb = x + ((size_t)(b * HH + yh) * WW) * CC + g * CG;
        // Hint the next tap row into cache while this one is consumed.
        if (i < KS - 1) {
            const int nh = hh + i - 2;
            if ((unsigned)nh < HH)
                __builtin_prefetch(x + ((size_t)(b * HH + nh) * WW + ww) * CC + g * CG, 0, 1);
        }
#pragma unroll
        for (int j = 0; j < KS; ++j) {
            const int yw = ww + j - 3;
            if ((unsigned)yw >= WW) continue;      // zero padding col
            const float wgt = wk[i * KS + j];
            const v4f* pv = (const v4f*)(rowb + (size_t)yw * CC);  // 16B aligned
            acc0 += wgt * pv[0];
            acc1 += wgt * pv[1];
            acc2 += wgt * pv[2];
            acc3 += wgt * pv[3];
        }
    }

    v4f* po = (v4f*)(out + (size_t)p * CC + g * CG);
    po[0] = acc0; po[1] = acc1; po[2] = acc2; po[3] = acc3;
}

// ===========================================================================
extern "C" void kernel_launch(void* const* d_in, const int* in_sizes, int n_in,
                              void* d_out, int out_size, void* d_ws, size_t ws_size,
                              hipStream_t stream) {
    const float* x        = (const float*)d_in[0];
    const float* w_reduce = (const float*)d_in[1];
    const float* gamma    = (const float*)d_in[2];
    const float* beta     = (const float*)d_in[3];
    const float* mean     = (const float*)d_in[4];
    const float* var      = (const float*)d_in[5];
    const float* w_span   = (const float*)d_in[6];
    const float* b_span   = (const float*)d_in[7];
    float* out = (float*)d_out;

    // Workspace: kgen f32 [P][784] first, then r bf16 [P][64].
    float* kgen_ws = (float*)d_ws;
    bf16*  r_ws    = (bf16*)((char*)d_ws + (size_t)PP * EE * sizeof(float));

    involution_reduce_bn<<<dim3(PP / 128), dim3(256), 0, stream>>>(
        x, w_reduce, gamma, beta, mean, var, r_ws);

    involution_span<<<dim3(PP / 128, 7), dim3(224), 0, stream>>>(
        r_ws, w_span, b_span, kgen_ws);

    involution_apply<<<dim3(PP * GG / 256), dim3(256), 0, stream>>>(
        x, kgen_ws, out);
}
